// CNN_binary_55001351193058
// MI455X (gfx1250) — 8-bit matrix operand order repaired and checked by reading, compiled, not run on hardware
//
#include <hip/hip_runtime.h>
#include <hip/hip_bf16.h>
#include <cstdint>
#include <cstddef>


typedef __attribute__((ext_vector_type(8))) int v8i;
typedef __attribute__((ext_vector_type(4))) int v4i;
typedef __attribute__((ext_vector_type(2))) int v2i;

static constexpr int BATCH = 8192;
static constexpr int HH    = 6;

static constexpr size_t alignup(size_t x) { return (x + 255) & ~size_t(255); }

static constexpr size_t OFF_W2B   = 0;
static constexpr size_t OFF_W3B   = alignup(OFF_W2B  + 64  * 96);
static constexpr size_t OFF_W4B   = alignup(OFF_W3B  + 128 * 192);
static constexpr size_t OFF_WFB   = alignup(OFF_W4B  + 128 * 768);
static constexpr size_t OFF_ACT1  = alignup(OFF_WFB  + 16  * 2048);
static constexpr size_t OFF_ACT2  = alignup(OFF_ACT1 + (size_t)BATCH*HH*32*32);
static constexpr size_t OFF_ACT3  = alignup(OFF_ACT2 + (size_t)BATCH*HH*32*64);
static constexpr size_t OFF_ACT3P = alignup(OFF_ACT3 + (size_t)BATCH*HH*32*128);
static constexpr size_t OFF_ACT4  = alignup(OFF_ACT3P+ (size_t)BATCH*HH*16*128);
static constexpr size_t WS_NEED   = OFF_ACT4 + (size_t)BATCH*2048;

__global__ __launch_bounds__(256) void prep_weights(
    const float* __restrict__ w2, const float* __restrict__ w3,
    const float* __restrict__ w4, const float* __restrict__ wf,
    signed char* __restrict__ w2b, signed char* __restrict__ w3b,
    signed char* __restrict__ w4b, signed char* __restrict__ wfb) {
  int i = blockIdx.x * 256 + threadIdx.x;
  if (i < 64 * 96) {
    int o = i / 96, k = i % 96, kw = k / 32, ci = k % 32;
    w2b[i] = (w2[(o * 32 + ci) * 3 + kw] >= 0.f) ? 1 : -1;
    return;
  }
  i -= 64 * 96;
  if (i < 128 * 192) {
    int o = i / 192, k = i % 192, kw = k / 64, ci = k % 64;
    w3b[i] = (w3[(o * 64 + ci) * 3 + kw] >= 0.f) ? 1 : -1;
    return;
  }
  i -= 128 * 192;
  if (i < 128 * 768) {
    int o = i / 768, k = i % 768, h = k / 128, ci = k % 128;
    w4b[i] = (w4[(o * 128 + ci) * 6 + h] >= 0.f) ? 1 : -1;
    return;
  }
  i -= 128 * 768;
  if (i < 16 * 2048) {
    int o = i / 2048, k = i % 2048;
    wfb[i] = (o < 10) ? ((wf[o * 2048 + k] >= 0.f) ? 1 : -1) : 0;
  }
}

__global__ __launch_bounds__(256) void conv1_bn_pool_bin(
    const float* __restrict__ x,  const float* __restrict__ w1,
    const float* __restrict__ b1, const float* __restrict__ g,
    const float* __restrict__ be, const float* __restrict__ mu,
    const float* __restrict__ vr, signed char* __restrict__ act1) {
  __shared__ float lw[288];
  __shared__ float lb[32], lsc[32], lsh[32];
  for (int i = threadIdx.x; i < 288; i += 256)
    lw[i] = (w1[i] >= 0.f) ? 1.f : -1.f;
  for (int i = threadIdx.x; i < 32; i += 256) {
    float inv = g[i] * rsqrtf(vr[i] + 1e-5f);
    lb[i] = b1[i]; lsc[i] = inv; lsh[i] = be[i] - mu[i] * inv;
  }
  __syncthreads();

  int idx = blockIdx.x * 256 + threadIdx.x;
  int b = idx / (HH * 32), rem = idx % (HH * 32);
  int h = rem >> 5, wp = rem & 31;
  const float* xr = x + ((size_t)b * HH + h) * 128;

  float xv[11];
  int xbase = 4 * wp - 4;
#pragma unroll
  for (int t = 0; t < 11; ++t) {
    int xi = xbase + t;
    xv[t] = (xi >= 0 && xi < 128) ? xr[xi] : 0.f;
  }
  signed char* outp = act1 + (size_t)idx * 32;
  for (int c = 0; c < 32; ++c) {
    float a0 = lb[c], a1 = lb[c];
#pragma unroll
    for (int t = 0; t < 9; ++t) {
      float wv = lw[c * 9 + t];
      a0 += xv[t] * wv;
      a1 += xv[t + 2] * wv;
    }
    float y = fmaxf(a0 * lsc[c] + lsh[c], a1 * lsc[c] + lsh[c]);
    outp[c] = (y >= 0.f) ? 1 : -1;
  }
}

template <int Cin, int Cout, int Ktot, int KPAD>
__global__ __launch_bounds__(128) void conv_w3_wmma(
    const signed char* __restrict__ act_in, const signed char* __restrict__ wts,
    const float* __restrict__ bias, const float* __restrict__ g,
    const float* __restrict__ be,   const float* __restrict__ mu,
    const float* __restrict__ vr,   signed char* __restrict__ act_out) {
  __shared__ signed char lwt[16 * KPAD];
  const int tid = threadIdx.x;
  const int n0  = blockIdx.y * 16;

  for (int i = tid; i < 16 * KPAD / 4; i += 128) {
    int byt = i * 4;
    int n = byt / KPAD, k = byt % KPAD;
    int val = (k < Ktot) ? *(const int*)(wts + (size_t)(n0 + n) * Ktot + k) : 0;
    *(int*)(lwt + byt) = val;
  }

  const int wave = tid >> 5, lane = tid & 31;
  const int lanehi = (lane >> 4) & 1, lanelo = lane & 15;
  const int m0 = (blockIdx.x * 4 + wave) * 16;

  const int n = n0 + lanelo;
  const float bi  = bias[n];
  const float inv = g[n] * rsqrtf(vr[n] + 1e-5f);
  const float sh  = be[n] - mu[n] * inv;

  const int m  = m0 + lanelo;
  const int bb = m / (HH * 32), rm = m % (HH * 32);
  const int h  = rm >> 5, w = rm & 31;
  const signed char* row1 = act_in + ((size_t)(bb * HH + h) * 32 + w) * Cin;
  const signed char* row0 = (w > 0)  ? row1 - Cin : row1;
  const signed char* row2 = (w < 31) ? row1 + Cin : row1;
  const int msk0 = (w > 0)  ? -1 : 0;
  const int msk2 = (w < 31) ? -1 : 0;

  __syncthreads();

  v8i accv[4];
#pragma unroll
  for (int q = 0; q < 4; ++q) accv[q] = v8i{0, 0, 0, 0, 0, 0, 0, 0};

#pragma unroll
  for (int ks = 0; ks < KPAD / 64; ++ks) {
    v8i va, vb;
#pragma unroll
    for (int c = 0; c < 4; ++c) {
      int a0 = 0, a1 = 0;
      if (ks * 64 + c * 16 < Ktot) {
        const int kg = ks * 64 + ((c >> 1) << 5) + ((c & 1) << 3) + (lanehi << 4);
        const int kw = kg / Cin;
        const int ci = kg & (Cin - 1);
        const signed char* p = (kw == 0) ? row0 : (kw == 1 ? row1 : row2);
        const int mk = (kw == 0) ? msk0 : (kw == 1 ? -1 : msk2);
        v2i t = *(const v2i*)(p + ci);
        a0 = t.x & mk; a1 = t.y & mk;
      }
      va[2 * c] = a0; va[2 * c + 1] = a1;
    }
    const signed char* lbp = lwt + lanelo * KPAD + ks * 64 + (lanehi << 4);
    v4i b0 = *(const v4i*)(lbp);
    v4i b1 = *(const v4i*)(lbp + 32);
    vb[0] = b0.x; vb[1] = b0.y; vb[2] = b0.z; vb[3] = b0.w;
    vb[4] = b1.x; vb[5] = b1.y; vb[6] = b1.z; vb[7] = b1.w;
    accv[ks & 3] = __builtin_amdgcn_wmma_i32_16x16x64_iu8(
        true, va, true, vb, accv[ks & 3], false, false);
  }
  v8i acc = (accv[0] + accv[1]) + (accv[2] + accv[3]);

#pragma unroll
  for (int q = 0; q < 8; ++q) {
    int mm = m0 + q + (lanehi << 3);
    int b2 = mm / (HH * 32), r2 = mm % (HH * 32);
    int h2 = r2 >> 5, w2 = r2 & 31;
    float f = ((float)acc[q] + bi) * inv + sh;
    act_out[((size_t)(b2 * HH + h2) * 32 + w2) * Cout + n] = (f >= 0.f) ? 1 : -1;
  }
}

__global__ __launch_bounds__(256) void pool3_bin(const v4i* __restrict__ a3,
                                                 v4i* __restrict__ a3p) {
  int i = blockIdx.x * 256 + threadIdx.x;
  int c = i & 7, wp = (i >> 3) & 15, hb = i >> 7;
  v4i x0 = a3[(size_t)(hb * 32 + 2 * wp) * 8 + c];
  v4i x1 = a3[(size_t)(hb * 32 + 2 * wp + 1) * 8 + c];
  a3p[i] = x0 & x1;
}

__global__ __launch_bounds__(128) void conv4_wmma(
    const signed char* __restrict__ act_in,
    const signed char* __restrict__ wts,
    const float* __restrict__ bias, const float* __restrict__ g,
    const float* __restrict__ be,   const float* __restrict__ mu,
    const float* __restrict__ vr,   signed char* __restrict__ act4) {
  __shared__ signed char lwt[16 * 768];
  const int tid = threadIdx.x;
  const int n0  = blockIdx.y * 16;
  for (int i = tid * 4; i < 16 * 768; i += 128 * 4)
    *(int*)(lwt + i) =
        *(const int*)(wts + (size_t)(n0 + i / 768) * 768 + (i % 768));

  const int wave = tid >> 5, lane = tid & 31;
  const int lanehi = (lane >> 4) & 1, lanelo = lane & 15;
  const int m0 = (blockIdx.x * 4 + wave) * 16;

  const int n = n0 + lanelo;
  const float bi  = bias[n];
  const float inv = g[n] * rsqrtf(vr[n] + 1e-5f);
  const float sh  = be[n] - mu[n] * inv;

  const int m  = m0 + lanelo;
  const int bb = m >> 4, w = m & 15;
  const signed char* inbw = act_in + (size_t)bb * (HH * 16 * 128) + w * 128;

  __syncthreads();

  v8i accv[4];
#pragma unroll
  for (int q = 0; q < 4; ++q) accv[q] = v8i{0, 0, 0, 0, 0, 0, 0, 0};

#pragma unroll
  for (int ks = 0; ks < 12; ++ks) {
    v8i va, vb;
#pragma unroll
    for (int c = 0; c < 4; ++c) {
      const int koff = ks * 64 + ((c >> 1) << 5) + ((c & 1) << 3) + (lanehi << 4);
      const int hh = koff >> 7;
      const int ci = (koff & 127);
      v2i t = *(const v2i*)(inbw + hh * (16 * 128) + ci);
      va[2 * c] = t.x; va[2 * c + 1] = t.y;
    }
    const signed char* lbp = lwt + lanelo * 768 + ks * 64 + (lanehi << 4);
    v4i b0 = *(const v4i*)(lbp);
    v4i b1 = *(const v4i*)(lbp + 32);
    vb[0] = b0.x; vb[1] = b0.y; vb[2] = b0.z; vb[3] = b0.w;
    vb[4] = b1.x; vb[5] = b1.y; vb[6] = b1.z; vb[7] = b1.w;
    accv[ks & 3] = __builtin_amdgcn_wmma_i32_16x16x64_iu8(
        true, va, true, vb, accv[ks & 3], false, false);
  }
  v8i acc = (accv[0] + accv[1]) + (accv[2] + accv[3]);

#pragma unroll
  for (int q = 0; q < 8; ++q) {
    int mm = m0 + q + (lanehi << 3);
    int b2 = mm >> 4, w2 = mm & 15;
    float f = ((float)acc[q] + bi) * inv + sh;
    act4[(size_t)b2 * 2048 + n * 16 + w2] = (f >= 0.f) ? 1 : -1;
  }
}

__global__ __launch_bounds__(128) void fc_wmma(
    const signed char* __restrict__ act4, const signed char* __restrict__ wfb,
    const float* __restrict__ bf, float* __restrict__ out) {
  __shared__ signed char lwt[16 * 2048];
  const int tid = threadIdx.x;
  for (int i = tid * 4; i < 16 * 2048; i += 128 * 4)
    *(int*)(lwt + i) = *(const int*)(wfb + i);

  const int wave = tid >> 5, lane = tid & 31;
  const int lanehi = (lane >> 4) & 1, lanelo = lane & 15;
  const int m0 = (blockIdx.x * 4 + wave) * 16;
  const signed char* arow = act4 + (size_t)(m0 + lanelo) * 2048;

  const int n = lanelo;
  const float bv = (n < 10) ? bf[n] : 0.f;

  __syncthreads();

  v8i accv[4];
#pragma unroll
  for (int q = 0; q < 4; ++q) accv[q] = v8i{0, 0, 0, 0, 0, 0, 0, 0};

#pragma unroll
  for (int ks = 0; ks < 32; ++ks) {
    v8i va, vb;
#pragma unroll
    for (int c = 0; c < 4; ++c) {
      v2i t = *(const v2i*)(arow + ks * 64 + ((c >> 1) << 5) + ((c & 1) << 3) + (lanehi << 4));
      va[2 * c] = t.x; va[2 * c + 1] = t.y;
    }
    const signed char* lbp = lwt + lanelo * 2048 + ks * 64 + (lanehi << 4);
    v4i b0 = *(const v4i*)(lbp);
    v4i b1 = *(const v4i*)(lbp + 32);
    vb[0] = b0.x; vb[1] = b0.y; vb[2] = b0.z; vb[3] = b0.w;
    vb[4] = b1.x; vb[5] = b1.y; vb[6] = b1.z; vb[7] = b1.w;
    accv[ks & 3] = __builtin_amdgcn_wmma_i32_16x16x64_iu8(
        true, va, true, vb, accv[ks & 3], false, false);
  }
  v8i acc = (accv[0] + accv[1]) + (accv[2] + accv[3]);

  if (n < 10) {
#pragma unroll
    for (int q = 0; q < 8; ++q) {
      int mm = m0 + q + (lanehi << 3);
      out[(size_t)mm * 10 + n] = (float)acc[q] + bv;
    }
  }
}

extern "C" void kernel_launch(void* const* d_in, const int* in_sizes, int n_in,
                              void* d_out, int out_size, void* d_ws,
                              size_t ws_size, hipStream_t stream) {
  (void)in_sizes; (void)n_in; (void)out_size;
  if (ws_size < WS_NEED) return;

  const float* x  = (const float*)d_in[0];
  const float* w1 = (const float*)d_in[1];  const float* b1 = (const float*)d_in[2];
  const float* w2 = (const float*)d_in[3];  const float* b2 = (const float*)d_in[4];
  const float* w3 = (const float*)d_in[5];  const float* b3 = (const float*)d_in[6];
  const float* w4 = (const float*)d_in[7];  const float* b4 = (const float*)d_in[8];
  const float* g1 = (const float*)d_in[9];  const float* be1 = (const float*)d_in[10];
  const float* m1 = (const float*)d_in[11]; const float* v1  = (const float*)d_in[12];
  const float* g2 = (const float*)d_in[13]; const float* be2 = (const float*)d_in[14];
  const float* m2 = (const float*)d_in[15]; const float* v2  = (const float*)d_in[16];
  const float* g3 = (const float*)d_in[17]; const float* be3 = (const float*)d_in[18];
  const float* m3 = (const float*)d_in[19]; const float* v3  = (const float*)d_in[20];
  const float* g4 = (const float*)d_in[21]; const float* be4 = (const float*)d_in[22];
  const float* m4 = (const float*)d_in[23]; const float* v4  = (const float*)d_in[24];
  const float* wf = (const float*)d_in[25]; const float* bf  = (const float*)d_in[26];

  char* ws = (char*)d_ws;
  signed char* w2b   = (signed char*)(ws + OFF_W2B);
  signed char* w3b   = (signed char*)(ws + OFF_W3B);
  signed char* w4b   = (signed char*)(ws + OFF_W4B);
  signed char* wfb   = (signed char*)(ws + OFF_WFB);
  signed char* act1  = (signed char*)(ws + OFF_ACT1);
  signed char* act2  = (signed char*)(ws + OFF_ACT2);
  signed char* act3  = (signed char*)(ws + OFF_ACT3);
  signed char* act3p = (signed char*)(ws + OFF_ACT3P);
  signed char* act4  = (signed char*)(ws + OFF_ACT4);
  float* out = (float*)d_out;

  prep_weights<<<632, 256, 0, stream>>>(w2, w3, w4, wf, w2b, w3b, w4b, wfb);

  conv1_bn_pool_bin<<<6144, 256, 0, stream>>>(x, w1, b1, g1, be1, m1, v1, act1);

  conv_w3_wmma<32, 64, 96, 128><<<dim3(24576, 4), 128, 0, stream>>>(
      act1, w2b, b2, g2, be2, m2, v2, act2);

  conv_w3_wmma<64, 128, 192, 192><<<dim3(24576, 8), 128, 0, stream>>>(
      act2, w3b, b3, g3, be3, m3, v3, act3);

  pool3_bin<<<24576, 256, 0, stream>>>((const v4i*)act3, (v4i*)act3p);

  conv4_wmma<<<dim3(2048, 8), 128, 0, stream>>>(
      act3p, w4b, b4, g4, be4, m4, v4, act4);

  fc_wmma<<<128, 128, 0, stream>>>(act4, wfb, bf, out);
}
